// LossFunction_86870008529070
// MI455X (gfx1250) — matrix operand order repaired and checked by reading, not compiled, not run on hardware
//
#include <hip/hip_runtime.h>
#include <math.h>

typedef __attribute__((ext_vector_type(16))) _Float16 v16h;
typedef __attribute__((ext_vector_type(8)))  _Float16 v8h;
typedef __attribute__((ext_vector_type(8)))  float    v8f;

#define B_ROWS   1024
#define DIM      256
#define NCLS     20000
#define KSC      3
#define WROWS    (NCLS * KSC)      // 60000
#define MBLK     4                 // M-tiles per wave (register blocking)

// margin constants (margin = 0.3, scale = 15)
#define C_COS_M  0.955336489125606f
#define C_SIN_M  0.2955202066613396f
#define C_TH    (-0.955336489125606f)        // cos(pi - m)
#define C_MM     0.08865606199840188f        // sin(pi - m) * m
#define C_SCALE  15.0f

// ---------------------------------------------------------------------------
// 1) L2-normalize rows of a [nrows x 256] f32 matrix, emit f16.
// ---------------------------------------------------------------------------
__global__ void __launch_bounds__(256) normalize_rows_f16(
    const float* __restrict__ in, _Float16* __restrict__ out) {
  __shared__ float red[256];
  const int row = blockIdx.x;
  const int tid = threadIdx.x;
  const float v = in[(size_t)row * DIM + tid];
  red[tid] = v * v;
  __syncthreads();
  for (int s = 128; s > 0; s >>= 1) {
    if (tid < s) red[tid] += red[tid + s];
    __syncthreads();
  }
  const float inv = 1.0f / fmaxf(sqrtf(red[0]), 1e-12f);
  out[(size_t)row * DIM + tid] = (_Float16)(v * inv);
}

// ---------------------------------------------------------------------------
// 2) Sub-center cosine GEMM with WMMA f16->f32, 4x M register blocking.
//    One wave32 computes a 64(M) x 16(class) block: the 3 sub-center B
//    fragments are loaded once per K-step and reused across 4 A tiles
//    (12 WMMAs per 14 b128 loads), cutting B-side L2 traffic 4x. Sub-center
//    max is fused before storing the [1024 x 20000] cosine table.
// ---------------------------------------------------------------------------
__global__ void __launch_bounds__(256) subcenter_cos_gemm(
    const _Float16* __restrict__ xn,      // [1024 x 256]
    const _Float16* __restrict__ wn,      // [60000 x 256]
    float* __restrict__ cosTab) {         // [1024 x 20000]
  const int lane   = threadIdx.x & 31;
  const int waveId = (blockIdx.x << 3) | (threadIdx.x >> 5);   // 20000 waves
  const int msup   = waveId & 15;          // 16 super-tiles of 64 rows
  const int ntile  = waveId >> 4;          // 1250 class tiles (20000/16)
  const int mbase  = msup << 6;
  const int nbase  = ntile << 4;
  const int l15    = lane & 15;
  const int khalf  = lane >> 4;

  // A fragment source rows (one per M-tile); K-halves selected by khalf.
  const _Float16* aRow = xn + (size_t)(mbase + l15) * DIM;
  // B fragment source: column l15 of this class tile -> weight row base.
  const _Float16* bRow = wn + (size_t)(nbase + l15) * KSC * DIM;

  v8f acc[MBLK][KSC] = {};

  for (int kb = 0; kb < DIM; kb += 32) {
    // A 16x32 f16 layout: lanes 0-15 hold K[kb..kb+15],
    // lanes 16-31 hold K[kb+16..kb+31] (matches B).
    v16h a[MBLK];
    #pragma unroll
    for (int i = 0; i < MBLK; ++i) {
      const _Float16* ap = aRow + (size_t)i * 16 * DIM;
      v8h alo = *(const v8h*)(ap + kb + khalf * 16);
      v8h ahi = *(const v8h*)(ap + kb + khalf * 16 + 8);
      a[i] = __builtin_shufflevector(alo, ahi,
          0, 1, 2, 3, 4, 5, 6, 7, 8, 9, 10, 11, 12, 13, 14, 15);
    }

    // B 32x16 f16 layout: lanes 0-15 hold K[kb..kb+15] of column l15,
    // lanes 16-31 hold K[kb+16..kb+31] (contiguous 32B per lane).
    const _Float16* bk = bRow + kb + khalf * 16;
    v16h b0 = *(const v16h*)(bk);
    v16h b1 = *(const v16h*)(bk + DIM);
    v16h b2 = *(const v16h*)(bk + 2 * DIM);

    #pragma unroll
    for (int i = 0; i < MBLK; ++i) {
      acc[i][0] = __builtin_amdgcn_wmma_f32_16x16x32_f16(false, a[i], false, b0,
                                              (short)0, acc[i][0], false, false);
      acc[i][1] = __builtin_amdgcn_wmma_f32_16x16x32_f16(false, a[i], false, b1,
                                              (short)0, acc[i][1], false, false);
      acc[i][2] = __builtin_amdgcn_wmma_f32_16x16x32_f16(false, a[i], false, b2,
                                              (short)0, acc[i][2], false, false);
    }
  }

  // C/D layout: element r -> (M = r + 8*khalf, N = l15). Fused sub-center max.
  #pragma unroll
  for (int i = 0; i < MBLK; ++i) {
    const size_t base =
        (size_t)(mbase + i * 16 + khalf * 8) * NCLS + (size_t)(nbase + l15);
    #pragma unroll
    for (int r = 0; r < 8; ++r) {
      const float m = fmaxf(fmaxf(acc[i][0][r], acc[i][1][r]), acc[i][2][r]);
      cosTab[base + (size_t)r * NCLS] = m;
    }
  }
}

// ---------------------------------------------------------------------------
// 3) Per-row margin + online log-sum-exp + argmax over 20000 classes.
//    Since mp==0, phi_mp == cosine and the top-k branch is a numeric no-op:
//    output[c] = SCALE * (c==target ? phi : cosine[c]).
// ---------------------------------------------------------------------------
__global__ void __launch_bounds__(256) row_loss_kernel(
    const float* __restrict__ cosTab, const int* __restrict__ label,
    float* __restrict__ rowLoss, float* __restrict__ rowCorrect) {
  __shared__ float sm[256], ss[256], sb[256];
  __shared__ int   si[256];
  const int b   = blockIdx.x;
  const int tid = threadIdx.x;
  const float* row = cosTab + (size_t)b * NCLS;
  const int tgt = label[b];

  const float cost = row[tgt];
  const float sint = sqrtf(fmaxf(1.0f - cost * cost, 0.0f));
  const float phi  = (cost - C_TH > 0.0f) ? (cost * C_COS_M - sint * C_SIN_M)
                                          : (cost - C_MM);
  const float outT = C_SCALE * phi;

  float m = -INFINITY, s = 0.0f, best = -INFINITY;
  int bidx = 0;
  for (int c = tid; c < NCLS; c += 256) {
    const float v = C_SCALE * ((c == tgt) ? phi : row[c]);
    if (v > best) { best = v; bidx = c; }
    if (v > m) { s = s * __expf(m - v) + 1.0f; m = v; }
    else       { s += __expf(v - m); }
  }
  sm[tid] = m; ss[tid] = s; sb[tid] = best; si[tid] = bidx;
  __syncthreads();
  for (int st = 128; st > 0; st >>= 1) {
    if (tid < st) {
      const float m2 = sm[tid + st], s2 = ss[tid + st];
      const float mx = fmaxf(sm[tid], m2);
      ss[tid] = ss[tid] * __expf(sm[tid] - mx) + s2 * __expf(m2 - mx);
      sm[tid] = mx;
      const float b2 = sb[tid + st];
      const int   i2 = si[tid + st];
      if (b2 > sb[tid] || (b2 == sb[tid] && i2 < si[tid])) {
        sb[tid] = b2; si[tid] = i2;
      }
    }
    __syncthreads();
  }
  if (tid == 0) {
    const float lse = sm[0] + __logf(ss[0]);
    rowLoss[b]    = -(outT - lse);
    rowCorrect[b] = (si[0] == tgt) ? 1.0f : 0.0f;
  }
}

// ---------------------------------------------------------------------------
// 4) Deterministic final reduction (no float atomics): mean loss + prec1.
// ---------------------------------------------------------------------------
__global__ void __launch_bounds__(256) finalize_kernel(
    const float* __restrict__ rowLoss, const float* __restrict__ rowCorrect,
    float* __restrict__ out) {
  __shared__ float sl[256], sc[256];
  const int tid = threadIdx.x;
  float l = 0.0f, c = 0.0f;
  for (int i = tid; i < B_ROWS; i += 256) { l += rowLoss[i]; c += rowCorrect[i]; }
  sl[tid] = l; sc[tid] = c;
  __syncthreads();
  for (int st = 128; st > 0; st >>= 1) {
    if (tid < st) { sl[tid] += sl[tid + st]; sc[tid] += sc[tid + st]; }
    __syncthreads();
  }
  if (tid == 0) {
    out[0] = sl[0] / (float)B_ROWS;
    out[1] = sc[0] / (float)B_ROWS * 100.0f;
  }
}

// ---------------------------------------------------------------------------
extern "C" void kernel_launch(void* const* d_in, const int* in_sizes, int n_in,
                              void* d_out, int out_size, void* d_ws, size_t ws_size,
                              hipStream_t stream) {
  const float* x     = (const float*)d_in[0];   // [1024 x 256]
  const float* w     = (const float*)d_in[1];   // [60000 x 256]
  const int*   label = (const int*)d_in[2];     // [1024]
  float* out = (float*)d_out;                   // [2] = (loss, prec1)

  // Workspace layout (base is 256B aligned from hipMalloc):
  char* ws = (char*)d_ws;
  _Float16* wn     = (_Float16*)ws;                                     // 30,720,000 B
  _Float16* xn     = (_Float16*)(ws + (size_t)WROWS * DIM * 2);         //    524,288 B
  float*    cosTab = (float*)(ws + (size_t)(WROWS + B_ROWS) * DIM * 2); // 81,920,000 B
  float*    rowLoss    = (float*)((char*)cosTab + (size_t)B_ROWS * NCLS * 4);
  float*    rowCorrect = rowLoss + B_ROWS;

  normalize_rows_f16<<<WROWS,  256, 0, stream>>>(w, wn);
  normalize_rows_f16<<<B_ROWS, 256, 0, stream>>>(x, xn);
  // 16 M-supertiles * 1250 class-tiles = 20000 waves = 2500 blocks of 8 waves
  subcenter_cos_gemm<<<2500, 256, 0, stream>>>(xn, wn, cosTab);
  row_loss_kernel<<<B_ROWS, 256, 0, stream>>>(cosTab, label, rowLoss, rowCorrect);
  finalize_kernel<<<1, 256, 0, stream>>>(rowLoss, rowCorrect, out);
}
